// CrossChannelAttention_25812753448994
// MI455X (gfx1250) — matrix operand order repaired and checked by reading, not compiled, not run on hardware
//
#include <hip/hip_runtime.h>

typedef __attribute__((ext_vector_type(16))) _Float16 v16h;
typedef __attribute__((ext_vector_type(8)))  _Float16 v8h;
typedef __attribute__((ext_vector_type(8)))  float    v8f;

#define BATCH 8
#define CH    128          // DIM
#define PIX   4096         // H*W
#define G     16           // NUMS
#define HEADS 8
#define OUTC  256
#define KDIM  2048         // G*CH
#define XT_STRIDE 136      // f16 [p][c] row stride: 17*16B -> b128-aligned, bank step 68%64=4

// ---------------------------------------------------------------------------
// Prep: W2 (f32 [256][2048]) -> f16, pre-swizzled into WMMA A-fragment order:
//   [mtile 0..15][kchunk 0..63][lane 0..31][elem 0..15]
// ISA 16-bit A 16x32 layout: lane holds row M = mtile*16 + (lane&15);
// koff = (lane>=16)?8:0; elems j<8 -> K=koff+j, j>=8 -> K=16+koff+(j-8).
// GEMM A-loads become coalesced 32B/lane global b128 pairs, no LDS, no barrier.
// ---------------------------------------------------------------------------
__global__ __launch_bounds__(256) void w2swz_kernel(const float* __restrict__ W2,
                                                    _Float16* __restrict__ w2s) {
  const int idx  = blockIdx.x * 256 + threadIdx.x;  // 0..32767 = 16*64*32
  const int lane = idx & 31;
  const int kc   = (idx >> 5) & 63;
  const int mt   = idx >> 11;
  const int m    = mt * 16 + (lane & 15);
  const int koff = (lane >> 4) * 8;
  v16h v;
#pragma unroll
  for (int j = 0; j < 16; ++j) {
    const int k = kc * 32 + ((j < 8) ? (koff + j) : (16 + koff + (j - 8)));
    v[j] = (_Float16)W2[(size_t)m * KDIM + k];
  }
  *(v16h*)(w2s + (size_t)idx * 16) = v;   // 32B-aligned, fully coalesced
}

// ---------------------------------------------------------------------------
// Fused kernel: per 128(M) x 128(N=p) output tile (256 threads = 8 waves):
//   1) stage x-tile transposed [p][c] as f16 in LDS (all 128 channels)
//   2) compute fm[g][p] = relu(b1[g] + sum_h xT[p][g*8+h]*W1[g,h]) in LDS
//   3) for each g: rebuild feat_g[p][c] = fm[g][p]*xT[p][c] (double-buffered,
//      one barrier per g), then 4 K-chunks: B-frags = 2x ds_load_b128 each,
//      A-frags = coalesced global v16h from pre-swizzled W2h (L2-resident),
//      8x v_wmma_f32_16x16x32_f16.
// Wave layout 4(Mw) x 2(Nw): each wave owns 32(M) x 64(N) = 8 f32 accumulators.
// ---------------------------------------------------------------------------
__global__ __launch_bounds__(256) void gemm_kernel(const float* __restrict__ x,
                                                   const float* __restrict__ W1,
                                                   const float* __restrict__ b1,
                                                   const _Float16* __restrict__ w2s,
                                                   const float* __restrict__ b2,
                                                   float* __restrict__ out) {
  __shared__ _Float16 xT[128 * XT_STRIDE];        // 34816 B  x-tile [p][c]
  __shared__ _Float16 ft[2][128 * XT_STRIDE];     // 69632 B  feat_g double buffer
  __shared__ _Float16 fmh[G][128];                //  4096 B
  __shared__ float    w1s[G * HEADS];
  __shared__ float    b1s[G];

  const int t     = threadIdx.x;
  const int b     = blockIdx.z;
  const int nBase = blockIdx.x * 128;
  const int mBlk  = blockIdx.y * 8;               // base mtile (8x16 rows = 128)

  if (t < G * HEADS) w1s[t] = W1[t];
  if (t < G)         b1s[t] = b1[t];

  // ---- stage x tile: coalesced global f32 reads -> transposed f16 LDS ----
  const float* xb = x + (size_t)b * CH * PIX + nBase;
#pragma unroll
  for (int i = 0; i < 64; ++i) {
    const int idx = t + i * 256;
    const int c = idx >> 7, p = idx & 127;
    xT[p * XT_STRIDE + c] = (_Float16)xb[(size_t)c * PIX + p];
  }
  __syncthreads();

  // ---- fm from staged tile (2048 values, 8 per thread) ----
#pragma unroll
  for (int i = 0; i < 8; ++i) {
    const int idx = t + i * 256;
    const int g = idx >> 7, p = idx & 127;
    const _Float16* xr = &xT[p * XT_STRIDE + g * HEADS];
    float s = b1s[g];
#pragma unroll
    for (int h = 0; h < HEADS; ++h) s += (float)xr[h] * w1s[g * HEADS + h];
    fmh[g][p] = (_Float16)fmaxf(s, 0.0f);
  }

  // ---- wave geometry ----
  const int wave = t >> 5, lane = t & 31;
  const int ln = lane & 15, hi = lane >> 4;
  const int mw = wave >> 1, nw = wave & 1;
  const int mt0  = mBlk + mw * 2;                 // wave's mtiles: mt0, mt0+1
  const int nOff = nw * 64;                       // wave's 64-pixel strip
  const int kb   = hi * 8;                        // per-half K base in chunk

  v8f acc[2][4] = {};
  const _Float16* aP0 = w2s + ((size_t)mt0 * 64 * 32 + lane) * 16;
  const _Float16* aP1 = aP0 + (size_t)64 * 32 * 16;

  // per-thread rebuild region: half a row of feat_g
  const int rp = t >> 1;
  const int rc = (t & 1) * 64;
  const _Float16* xrow = &xT[rp * XT_STRIDE + rc];

  __syncthreads();                                // fmh visible

  // prologue: build feat for g=0
  {
    const _Float16 fv = fmh[0][rp];
    _Float16* frow = &ft[0][rp * XT_STRIDE + rc];
#pragma unroll
    for (int i = 0; i < 8; ++i) {
      v8h xv = *(const v8h*)(xrow + i * 8);
      v8h pr;
#pragma unroll
      for (int j = 0; j < 8; ++j) pr[j] = xv[j] * fv;
      *(v8h*)(frow + i * 8) = pr;
    }
  }
  __syncthreads();

  for (int g = 0; g < G; ++g) {
    // build feat for g+1 into the idle buffer (no barrier needed: that buffer
    // was last read before the barrier ending iteration g-1)
    if (g + 1 < G) {
      const _Float16 fv = fmh[g + 1][rp];
      _Float16* frow = &ft[(g + 1) & 1][rp * XT_STRIDE + rc];
#pragma unroll
      for (int i = 0; i < 8; ++i) {
        v8h xv = *(const v8h*)(xrow + i * 8);
        v8h pr;
#pragma unroll
        for (int j = 0; j < 8; ++j) pr[j] = xv[j] * fv;
        *(v8h*)(frow + i * 8) = pr;
      }
      // warm L0 for next g's A stream (emits global_prefetch_b8; L2-resident)
      __builtin_prefetch(aP0 + (size_t)(g + 1) * 4 * 512, 0, 3);
      __builtin_prefetch(aP1 + (size_t)(g + 1) * 4 * 512, 0, 3);
    }

    const _Float16* fcur = ft[g & 1];
#pragma unroll
    for (int q = 0; q < 4; ++q) {
      const int kc = g * 4 + q;
      const v16h a0 = *(const v16h*)(aP0 + (size_t)kc * 512);   // 512 = 32 lanes * 16
      const v16h a1 = *(const v16h*)(aP1 + (size_t)kc * 512);
#pragma unroll
      for (int nt = 0; nt < 4; ++nt) {
        const int p = nOff + nt * 16 + ln;
        union { v16h v; v8h h[2]; } bf;                         // K: kb..kb+7, 16+kb..16+kb+7
        const _Float16* fr = &fcur[p * XT_STRIDE + q * 32 + kb];
        bf.h[0] = *(const v8h*)fr;                              // ds_load_b128
        bf.h[1] = *(const v8h*)(fr + 16);                       // ds_load_b128
        acc[0][nt] = __builtin_amdgcn_wmma_f32_16x16x32_f16(
            false, a0, false, bf.v, (short)0, acc[0][nt], false, false);
        acc[1][nt] = __builtin_amdgcn_wmma_f32_16x16x32_f16(
            false, a1, false, bf.v, (short)0, acc[1][nt], false, false);
      }
    }
    __syncthreads();   // feat(g) consumed by all; feat(g+1) build visible
  }

  // ---- epilogue: C/D layout -> VGPR r holds M = r + 8*hi; lane ln = col p ----
  float* ob = out + (size_t)b * OUTC * PIX;
#pragma unroll
  for (int mi = 0; mi < 2; ++mi) {
    const int obase = (mt0 + mi) * 16 + hi * 8;
#pragma unroll
    for (int nt = 0; nt < 4; ++nt) {
      const int p = nBase + nOff + nt * 16 + ln;
#pragma unroll
      for (int r = 0; r < 8; ++r) {
        const int o = obase + r;
        ob[(size_t)o * PIX + p] = acc[mi][nt][r] + b2[o];
      }
    }
  }
}

// ---------------------------------------------------------------------------
extern "C" void kernel_launch(void* const* d_in, const int* in_sizes, int n_in,
                              void* d_out, int out_size, void* d_ws, size_t ws_size,
                              hipStream_t stream) {
  const float* x  = (const float*)d_in[0];
  const float* W1 = (const float*)d_in[1];
  const float* b1 = (const float*)d_in[2];
  const float* W2 = (const float*)d_in[3];
  const float* b2 = (const float*)d_in[4];
  float* out = (float*)d_out;

  _Float16* w2s = (_Float16*)d_ws;   // 1 MB swizzled f16 W2

  w2swz_kernel<<<(16 * 64 * 32) / 256, 256, 0, stream>>>(W2, w2s);
  dim3 grid(PIX / 128, OUTC / 128, BATCH);
  gemm_kernel<<<grid, 256, 0, stream>>>(x, W1, b1, w2s, b2, out);
}
